// LSTM_Hedging_180388626578
// MI455X (gfx1250) — matrix operand order repaired and checked by reading, not compiled, not run on hardware
//
#include <hip/hip_runtime.h>

#define B_SZ 16384
#define T_SZ 512
#define H_SZ 128
#define TP1  513

typedef __attribute__((ext_vector_type(16))) __bf16 v16bf;
typedef __attribute__((ext_vector_type(8)))  __bf16 v8bf;
typedef __attribute__((ext_vector_type(8)))  float  v8f;

// ---- workspace layout (bytes) ----
#define WS_W1    0          // [512 n][256 k] bf16  (k<128: W_ih1, k>=128: W_hh1)  = 262144
#define WS_WHH0  262144     // [512 n][128 k] bf16                                  = 131072
#define WS_W0    393216     // [512 n][ 32 k] bf16  (k<4: W_ih0 incl delta col)     = 32768
#define WS_B0    425984     // 512 f32  (b_ih0 + b_hh0)
#define WS_B1    428032     // 512 f32  (b_ih1 + b_hh1)

// ---- LDS layout (bytes) ----
#define LDS_W1   0          // 262144
#define LDS_H0   262144     // 16 x 128 bf16 = 4096
#define LDS_H1   266240     // 4096
#define LDS_DLT  270336     // 16 f32 = 64
#define LDS_TOT  270400

#if defined(__has_builtin)
#if __has_builtin(__builtin_amdgcn_tanhf)
#define HW_TANH 1
#endif
#endif

// sigmoid(x + b) ; bias folded into the FMA so it costs nothing extra
__device__ __forceinline__ float act_sigmoid(float x, float b) {
#ifdef HW_TANH
    return 0.5f + 0.5f * __builtin_amdgcn_tanhf(0.5f * x + 0.5f * b);
#else
    return __builtin_amdgcn_rcpf(
        1.0f + __builtin_amdgcn_exp2f(-1.4426950408889634f * x +
                                      (-1.4426950408889634f) * b));
#endif
}
// tanh(x + b)
__device__ __forceinline__ float act_tanh(float x, float b) {
#ifdef HW_TANH
    return __builtin_amdgcn_tanhf(x + b);
#else
    return 2.0f * __builtin_amdgcn_rcpf(
                      1.0f + __builtin_amdgcn_exp2f(
                                 -2.8853900817779268f * x +
                                 (-2.8853900817779268f) * b)) -
           1.0f;
#endif
}

union V16U { v16bf v; v8bf h[2]; };

// A-fragment (16x32 bf16): lane L holds M = L&15; elems 0-15 = K[k0+half*16 ..)
// (contiguous, matching load_bfrag)
__device__ __forceinline__ v16bf load_afrag(const __bf16* base, int rowStride,
                                            int m, int k0, int kh) {
    V16U u;
    const __bf16* p = base + m * rowStride + k0 + kh * 16;
    u.h[0] = *(const v8bf*)(p);
    u.h[1] = *(const v8bf*)(p + 8);
    return u.v;
}

// B-fragment (32x16 bf16): lane L holds N column n0+(L&15); 16 contiguous K
// starting at k0 + half*16  (same contiguous k order as load_afrag)
__device__ __forceinline__ v16bf load_bfrag(const __bf16* base, int rowStride,
                                            int n, int k0, int kh) {
    const __bf16* p = base + n * rowStride + k0 + kh * 16;
    return *(const v16bf*)(p);
}

// ------------------------------------------------------------------
// Prep: f32 -> bf16 weight repack + bias folding into workspace
// ------------------------------------------------------------------
__global__ void prep_kernel(const float* Wih0, const float* Whh0,
                            const float* bih0, const float* bhh0,
                            const float* Wih1, const float* Whh1,
                            const float* bih1, const float* bhh1,
                            char* ws) {
    int idx = blockIdx.x * 256 + threadIdx.x;
    __bf16* w1   = (__bf16*)(ws + WS_W1);
    __bf16* whh0 = (__bf16*)(ws + WS_WHH0);
    __bf16* w0   = (__bf16*)(ws + WS_W0);
    float*  b0   = (float*)(ws + WS_B0);
    float*  b1   = (float*)(ws + WS_B1);
    if (idx < 512 * 256) {
        int n = idx >> 8, k = idx & 255;
        float v = (k < 128) ? Wih1[n * 128 + k] : Whh1[n * 128 + (k - 128)];
        w1[n * 256 + k] = (__bf16)v;
    }
    if (idx < 512 * 128) {
        int n = idx >> 7, k = idx & 127;
        whh0[n * 128 + k] = (__bf16)Whh0[n * 128 + k];
    }
    if (idx < 512 * 32) {
        int n = idx >> 5, k = idx & 31;
        w0[n * 32 + k] = (__bf16)((k < 4) ? Wih0[n * 4 + k] : 0.0f);
    }
    if (idx < 512) {
        b0[idx] = bih0[idx] + bhh0[idx];
        b1[idx] = bih1[idx] + bhh1[idx];
    }
}

// ------------------------------------------------------------------
// Main persistent LSTM kernel: 1 WG = 16 batch rows, 8 waves partition
// the 512 gate columns (wave w owns hidden units [16w,16w+16) per gate)
// ------------------------------------------------------------------
__global__ void __launch_bounds__(256, 1)
lstm_kernel(const float* __restrict__ data, const float* __restrict__ PnL,
            const float* __restrict__ fcW, const float* __restrict__ fcb,
            const char* __restrict__ ws, float* __restrict__ out) {
    extern __shared__ char smem[];
    __bf16* w1L  = (__bf16*)(smem + LDS_W1);
    __bf16* h0L  = (__bf16*)(smem + LDS_H0);
    __bf16* h1L  = (__bf16*)(smem + LDS_H1);
    float*  dltL = (float*)(smem + LDS_DLT);

    const int tid    = threadIdx.x;
    const int wv     = tid >> 5;
    const int ln     = tid & 31;
    const int lane16 = ln & 15;
    const int kh     = ln >> 4;
    const int m0     = blockIdx.x * 16;

    // stage layer-1 weights (bf16) into LDS: 256 KB
    {
        const uint4* src = (const uint4*)(ws + WS_W1);
        uint4*       dst = (uint4*)w1L;
        for (int i = tid; i < 262144 / 16; i += 256) dst[i] = src[i];
    }
    // zero h0/h1 state buffers (contiguous 8192 B) and delta
    for (int i = tid; i < 8192 / 4; i += 256) ((unsigned*)h0L)[i] = 0u;
    if (tid < 16) dltL[tid] = 0.0f;

    // persistent per-wave register weights: W_hh0 slice + padded W_ih0 slice
    const __bf16* whh0G = (const __bf16*)(ws + WS_WHH0);
    const __bf16* w0G   = (const __bf16*)(ws + WS_W0);
    const float*  b0G   = (const float*)(ws + WS_B0);
    const float*  b1G   = (const float*)(ws + WS_B1);

    v16bf whh0f[4][4];
    v16bf w0f[4];
    float bias0[4], bias1[4];
#pragma unroll
    for (int g = 0; g < 4; ++g) {
        int n = g * 128 + wv * 16 + lane16;
#pragma unroll
        for (int kt = 0; kt < 4; ++kt)
            whh0f[g][kt] = load_bfrag(whh0G, 128, n, kt * 32, kh);
        w0f[g]   = load_bfrag(w0G, 32, n, 0, kh);
        bias0[g] = b0G[n];
        bias1[g] = b1G[n];
    }

    const v8f vzero = {};   // C = inline 0 for first WMMA of each gate chain
    v8f c0 = {};
    v8f c1 = {};
    float delta_prev = 0.0f, pnl = 0.0f;

    // fc-head thread mapping: 16 threads per batch row
    const int r = tid >> 4;   // batch row within tile
    const int s = tid & 15;   // 8-element segment of the 128-dot
    float fw[8];
#pragma unroll
    for (int j = 0; j < 8; ++j) fw[j] = fcW[s * 8 + j];
    if (s == 0) out[(size_t)(m0 + r) * TP1] = 0.0f;   // delta_his[:,0] = 0

    __syncthreads();

    for (int t = 0; t < T_SZ; ++t) {
        // ---- build layer-0 input A-fragment: [x0,x1,x2,delta,0...] ----
        v16bf xa;
#pragma unroll
        for (int e = 0; e < 16; ++e) xa[e] = (__bf16)0.0f;
        if (kh == 0) {
            const float* dp = data + ((size_t)(m0 + lane16) * T_SZ + t) * 3;
            xa[0] = (__bf16)dp[0];
            xa[1] = (__bf16)dp[1];
            xa[2] = (__bf16)dp[2];
            xa[3] = (__bf16)dltL[lane16];
            __builtin_prefetch((const void*)(dp + 48), 0, 3);  // ~16 steps ahead
        }
        // ---- layer 0: gates = xA@W0 + h0@Whh0  (bias applied in activation) ----
        v16bf a0[4];
#pragma unroll
        for (int kt = 0; kt < 4; ++kt)
            a0[kt] = load_afrag(h0L, 128, lane16, kt * 32, kh);

        v8f acc[4];
#pragma unroll
        for (int g = 0; g < 4; ++g) {
            acc[g] = __builtin_amdgcn_wmma_f32_16x16x32_bf16(
                false, xa, false, w0f[g], (short)0, vzero, false, false);
#pragma unroll
            for (int kt = 0; kt < 4; ++kt)
                acc[g] = __builtin_amdgcn_wmma_f32_16x16x32_bf16(
                    false, a0[kt], false, whh0f[g][kt], (short)0, acc[g],
                    false, false);
        }
        v8f hnew0 = {};
#pragma unroll
        for (int e = 0; e < 8; ++e) {
            float gi = act_sigmoid(acc[0][e], bias0[0]);
            float gf = act_sigmoid(acc[1][e], bias0[1]);
            float gg = act_tanh(acc[2][e], bias0[2]);
            float go = act_sigmoid(acc[3][e], bias0[3]);
            float cc = gf * c0[e] + gi * gg;
            c0[e]    = cc;
            hnew0[e] = go * act_tanh(cc, 0.0f);
        }
        __syncthreads();   // all waves done reading old h0
        {
            int col = wv * 16 + lane16;   // hidden unit owned by this wave
#pragma unroll
            for (int e = 0; e < 8; ++e)
                h0L[(e + kh * 8) * 128 + col] = (__bf16)hnew0[e];
        }
        __syncthreads();

        // ---- layer 1: gates = [h0_new ; h1_old] @ W1  (K=256) ----
        v16bf a1[8];
#pragma unroll
        for (int kt = 0; kt < 4; ++kt)
            a1[kt] = load_afrag(h0L, 128, lane16, kt * 32, kh);
#pragma unroll
        for (int kt = 0; kt < 4; ++kt)
            a1[4 + kt] = load_afrag(h1L, 128, lane16, kt * 32, kh);

#pragma unroll
        for (int g = 0; g < 4; ++g) {
            int n = g * 128 + wv * 16 + lane16;
            {
                v16bf bf = load_bfrag(w1L, 256, n, 0, kh);
                acc[g] = __builtin_amdgcn_wmma_f32_16x16x32_bf16(
                    false, a1[0], false, bf, (short)0, vzero, false, false);
            }
#pragma unroll
            for (int kt = 1; kt < 8; ++kt) {
                v16bf bf = load_bfrag(w1L, 256, n, kt * 32, kh);
                acc[g] = __builtin_amdgcn_wmma_f32_16x16x32_bf16(
                    false, a1[kt], false, bf, (short)0, acc[g], false, false);
            }
        }
        v8f hnew1 = {};
#pragma unroll
        for (int e = 0; e < 8; ++e) {
            float gi = act_sigmoid(acc[0][e], bias1[0]);
            float gf = act_sigmoid(acc[1][e], bias1[1]);
            float gg = act_tanh(acc[2][e], bias1[2]);
            float go = act_sigmoid(acc[3][e], bias1[3]);
            float cc = gf * c1[e] + gi * gg;
            c1[e]    = cc;
            hnew1[e] = go * act_tanh(cc, 0.0f);
        }
        __syncthreads();   // all waves done reading old h1
        {
            int col = wv * 16 + lane16;
#pragma unroll
            for (int e = 0; e < 8; ++e)
                h1L[(e + kh * 8) * 128 + col] = (__bf16)hnew1[e];
        }
        __syncthreads();

        // ---- fc head: delta_new = h1 . fcW + fcb ; pnl recurrence ----
        float part = 0.0f;
#pragma unroll
        for (int j = 0; j < 8; ++j)
            part += (float)h1L[r * 128 + s * 8 + j] * fw[j];
#pragma unroll
        for (int off = 8; off >= 1; off >>= 1)
            part += __shfl_xor(part, off, 16);
        if (s == 0) {
            float dnew  = part + fcb[0];
            float price = data[((size_t)(m0 + r) * T_SZ + t) * 3];
            pnl += (delta_prev - dnew) * price;
            delta_prev = dnew;
            dltL[r]    = dnew;
            out[(size_t)(m0 + r) * TP1 + t + 1] = dnew;
        }
        __syncthreads();   // dltL visible for next step's xA build
    }

    if (s == 0) {
        float lastp = data[((size_t)(m0 + r) * T_SZ + (T_SZ - 1)) * 3];
        float tot   = pnl + delta_prev * lastp + PnL[m0 + r];
        out[(size_t)B_SZ * TP1 + m0 + r] = tot;
    }
}

// ------------------------------------------------------------------
extern "C" void kernel_launch(void* const* d_in, const int* in_sizes, int n_in,
                              void* d_out, int out_size, void* d_ws,
                              size_t ws_size, hipStream_t stream) {
    const float* data = (const float*)d_in[0];
    const float* PnLp = (const float*)d_in[1];
    const float* Wih0 = (const float*)d_in[2];
    const float* Whh0 = (const float*)d_in[3];
    const float* bih0 = (const float*)d_in[4];
    const float* bhh0 = (const float*)d_in[5];
    const float* Wih1 = (const float*)d_in[6];
    const float* Whh1 = (const float*)d_in[7];
    const float* bih1 = (const float*)d_in[8];
    const float* bhh1 = (const float*)d_in[9];
    const float* fcW  = (const float*)d_in[10];
    const float* fcb  = (const float*)d_in[11];
    // d_in[12] = price_loc (== 0 per reference setup)
    char*  ws  = (char*)d_ws;
    float* out = (float*)d_out;

    prep_kernel<<<512, 256, 0, stream>>>(Wih0, Whh0, bih0, bhh0,
                                         Wih1, Whh1, bih1, bhh1, ws);
    lstm_kernel<<<B_SZ / 16, 256, LDS_TOT, stream>>>(data, PnLp, fcW, fcb,
                                                     ws, out);
}
